// GPTBigCodeAttention_70119636074915
// MI455X (gfx1250) — matrix operand order repaired and checked by reading, not compiled, not run on hardware
//
#include <hip/hip_runtime.h>

// ---------------------------------------------------------------------------
// Types (clang ext-vectors only: trivially constructible, union-safe)
// ---------------------------------------------------------------------------
typedef __bf16 bf16;
typedef __attribute__((ext_vector_type(16))) __bf16 v16bf;
typedef __attribute__((ext_vector_type(8)))  float  v8f;
typedef __attribute__((ext_vector_type(4)))  unsigned u32x4;
typedef __attribute__((ext_vector_type(2)))  unsigned u32x2;
typedef __attribute__((ext_vector_type(4)))  float  f32x4;

union Frag {           // one 16x32 (A) or 32x16 (B) bf16 fragment: 8 VGPRs
    v16bf v;
    u32x4 q[2];
};

static __device__ __forceinline__ bf16 f2bf(float f) {
    unsigned u = __builtin_bit_cast(unsigned, f);
    unsigned r = u + 0x7FFFu + ((u >> 16) & 1u);   // round-to-nearest-even
    unsigned short h = (unsigned short)(r >> 16);
    return __builtin_bit_cast(bf16, h);
}

static __device__ __forceinline__ v8f wmma_bf16(const Frag& a, const Frag& b, v8f c) {
    // D = A(16x32) x B(32x16) + C, fp32 accumulate
    return __builtin_amdgcn_wmma_f32_16x16x32_bf16(
        false, a.v, false, b.v, (short)0, c, false, false);
}

// ---------------------------------------------------------------------------
// Constants for this problem
// ---------------------------------------------------------------------------
#define HID   2048       // hidden size
#define NHEAD 16
#define HD    128
#define SEQ   2048
#define BATCH 2
#define MROWS (BATCH*SEQ)          // 4096
#define NQKV  (HID + 2*HD)         // 2304
#define QSCALE 0.08838834764831845f   // 128^-0.5

// ---------------------------------------------------------------------------
// Kernel 1: fp32 -> bf16 elementwise (x)
// ---------------------------------------------------------------------------
__global__ void cvt_bf16_kernel(const float* __restrict__ in,
                                bf16* __restrict__ out, int n4) {
    int i = blockIdx.x * blockDim.x + threadIdx.x;
    if (i >= n4) return;
    f32x4 f = ((const f32x4*)in)[i];
    unsigned short h0 = __builtin_bit_cast(unsigned short, f2bf(f.x));
    unsigned short h1 = __builtin_bit_cast(unsigned short, f2bf(f.y));
    unsigned short h2 = __builtin_bit_cast(unsigned short, f2bf(f.z));
    unsigned short h3 = __builtin_bit_cast(unsigned short, f2bf(f.w));
    u32x2 p;
    p.x = (unsigned)h0 | ((unsigned)h1 << 16);
    p.y = (unsigned)h2 | ((unsigned)h3 << 16);
    *(u32x2*)(out + 4 * (size_t)i) = p;
}

// ---------------------------------------------------------------------------
// Kernel 2: fp32 [R,C] -> bf16 transposed [C,R] (weights), LDS-tiled
// ---------------------------------------------------------------------------
__global__ void transpose_cvt_kernel(const float* __restrict__ in,
                                     bf16* __restrict__ out, int R, int C) {
    __shared__ float tile[32][33];
    int r0 = blockIdx.x * 32, c0 = blockIdx.y * 32;
    int tx = threadIdx.x, ty = threadIdx.y;   // block (32,8)
#pragma unroll
    for (int i = 0; i < 32; i += 8)
        tile[ty + i][tx] = in[(size_t)(r0 + ty + i) * C + c0 + tx];
    __syncthreads();
#pragma unroll
    for (int i = 0; i < 32; i += 8)
        out[(size_t)(c0 + ty + i) * R + r0 + tx] = f2bf(tile[tx][ty + i]);
}

// ---------------------------------------------------------------------------
// Kernel 3: bf16 WMMA GEMM  C[M,N] = A[M,K] * BT[N,K]^T + bias
//   block = 256 threads = 8 waves; block tile 128x128; wave tile 32x64
//   mode 0: QKV epilogue (split q*scale / k / v-transposed, all bf16)
//   mode 1: proj epilogue (fp32 out)
// ---------------------------------------------------------------------------
__launch_bounds__(256)
__global__ void gemm_wmma_kernel(const bf16* __restrict__ A,
                                 const bf16* __restrict__ BT,
                                 const float* __restrict__ bias,
                                 int K, int mode,
                                 bf16* __restrict__ qout,
                                 bf16* __restrict__ kout,
                                 bf16* __restrict__ vtout,
                                 float* __restrict__ fout) {
    const int lane = threadIdx.x & 31;
    const int wave = threadIdx.x >> 5;
    const int wm = wave & 3, wn = wave >> 2;
    const int m0 = blockIdx.x * 128 + wm * 32;
    const int n0 = blockIdx.y * 128 + wn * 64;
    const int ln = lane & 15, hi = lane >> 4;

    v8f acc[2][4] = {};

    for (int k0 = 0; k0 < K; k0 += 32) {
        Frag a[2], b[4];
#pragma unroll
        for (int mt = 0; mt < 2; ++mt) {
            const bf16* p = A + (size_t)(m0 + mt * 16 + ln) * K + k0 + 16 * hi;
            a[mt].q[0] = *(const u32x4*)p;
            a[mt].q[1] = *(const u32x4*)(p + 8);
            __builtin_prefetch(p + 32, 0, 1);     // next k-tile -> global_prefetch
        }
#pragma unroll
        for (int nt = 0; nt < 4; ++nt) {
            const bf16* p = BT + (size_t)(n0 + nt * 16 + ln) * K + k0 + 16 * hi;
            b[nt].q[0] = *(const u32x4*)p;
            b[nt].q[1] = *(const u32x4*)(p + 8);
            __builtin_prefetch(p + 32, 0, 1);
        }
#pragma unroll
        for (int mt = 0; mt < 2; ++mt)
#pragma unroll
            for (int nt = 0; nt < 4; ++nt)
                acc[mt][nt] = wmma_bf16(a[mt], b[nt], acc[mt][nt]);
    }

    // epilogue: C layout -> element (M = r + 8*hi, N = ln) in VGPR r
#pragma unroll
    for (int mt = 0; mt < 2; ++mt) {
#pragma unroll
        for (int nt = 0; nt < 4; ++nt) {
#pragma unroll
            for (int r = 0; r < 8; ++r) {
                int m = m0 + mt * 16 + r + 8 * hi;
                int n = n0 + nt * 16 + ln;
                float val = acc[mt][nt][r] + bias[n];
                if (mode == 0) {
                    if (n < HID) {
                        qout[(size_t)m * HID + n] = f2bf(val * QSCALE);
                    } else if (n < HID + HD) {
                        kout[(size_t)m * HD + (n - HID)] = f2bf(val);
                    } else {
                        int d = n - (HID + HD);
                        int bb = m >> 11, s = m & (SEQ - 1);
                        vtout[((size_t)bb * HD + d) * SEQ + s] = f2bf(val);
                    }
                } else {
                    fout[(size_t)m * HID + n] = val;
                }
            }
        }
    }
}

// ---------------------------------------------------------------------------
// Kernel 4: causal MQA flash attention, one wave per (b, head, 16 q rows)
//   scores tile 16x32 per chunk: 8 WMMA (QK^T) + 8 WMMA (P·V)
//   P re-striped C-layout -> A-layout through 1KB LDS
// ---------------------------------------------------------------------------
__launch_bounds__(32)
__global__ void mqa_attn_kernel(const bf16* __restrict__ Q,
                                const bf16* __restrict__ Kb,
                                const bf16* __restrict__ Vt,
                                bf16* __restrict__ O) {
    const int qb = blockIdx.x * 16;
    const int h  = blockIdx.y;
    const int b  = blockIdx.z;
    const int lane = threadIdx.x;
    const int ln = lane & 15, hi = lane >> 4;

    __shared__ __align__(16) bf16 Plds[16 * 32];

    // preload Q A-fragments for d = 0..127 (4 K-steps of 32)
    Frag qf[4];
    {
        const bf16* qrow = Q + (size_t)(b * SEQ + qb + ln) * HID + h * HD;
#pragma unroll
        for (int t = 0; t < 4; ++t) {
            qf[t].q[0] = *(const u32x4*)(qrow + t * 32 + 16 * hi);
            qf[t].q[1] = *(const u32x4*)(qrow + t * 32 + 8 + 16 * hi);
        }
    }

    v8f oacc[8] = {};
    float mrow[8], lrow[8];
#pragma unroll
    for (int r = 0; r < 8; ++r) { mrow[r] = -1e30f; lrow[r] = 0.0f; }

    const bf16* kb = Kb + (size_t)b * SEQ * HD;
    const bf16* vb = Vt + (size_t)b * HD * SEQ;

    const int nkv = qb + 16;
    for (int kv0 = 0; kv0 < nkv; kv0 += 32) {
        // ---- scores = (Q*scale) . K^T : two 16x16 n-tiles
        v8f s0 = {}, s1 = {};
#pragma unroll
        for (int t = 0; t < 4; ++t) {
            Frag kf0, kf1;
            const bf16* p0 = kb + (size_t)(kv0 + ln) * HD + t * 32 + 16 * hi;
            kf0.q[0] = *(const u32x4*)p0;
            kf0.q[1] = *(const u32x4*)(p0 + 8);
            const bf16* p1 = kb + (size_t)(kv0 + 16 + ln) * HD + t * 32 + 16 * hi;
            kf1.q[0] = *(const u32x4*)p1;
            kf1.q[1] = *(const u32x4*)(p1 + 8);
            s0 = wmma_bf16(qf[t], kf0, s0);
            s1 = wmma_bf16(qf[t], kf1, s1);
        }

        // ---- causal mask (only chunks touching the diagonal)
        if (kv0 + 31 > qb) {
#pragma unroll
            for (int r = 0; r < 8; ++r) {
                int mg = qb + r + 8 * hi;
                if (kv0 + ln > mg)       s0[r] = -1e30f;
                if (kv0 + 16 + ln > mg)  s1[r] = -1e30f;
            }
        }

        // ---- online softmax (row = r + 8*hi, reduce across own 16-lane half)
        float p0v[8], p1v[8], alpha[8];
#pragma unroll
        for (int r = 0; r < 8; ++r) {
            float mx = fmaxf(s0[r], s1[r]);
#pragma unroll
            for (int off = 8; off; off >>= 1)
                mx = fmaxf(mx, __shfl_xor(mx, off, 32));
            float mnew = fmaxf(mrow[r], mx);
            alpha[r] = __expf(mrow[r] - mnew);
            mrow[r] = mnew;
            p0v[r] = __expf(s0[r] - mnew);
            p1v[r] = __expf(s1[r] - mnew);
            float sum = p0v[r] + p1v[r];
#pragma unroll
            for (int off = 8; off; off >>= 1)
                sum += __shfl_xor(sum, off, 32);
            lrow[r] = lrow[r] * alpha[r] + sum;
        }

        // rescale running output
#pragma unroll
        for (int dt = 0; dt < 8; ++dt)
#pragma unroll
            for (int r = 0; r < 8; ++r)
                oacc[dt][r] *= alpha[r];

        // ---- P: C-layout -> LDS (row-major 16x32 bf16) -> A-layout
#pragma unroll
        for (int r = 0; r < 8; ++r) {
            int row = r + 8 * hi;
            Plds[row * 32 + ln]      = f2bf(p0v[r]);
            Plds[row * 32 + 16 + ln] = f2bf(p1v[r]);
        }
        __syncthreads();
        Frag pf;
        {
            const bf16* pr = &Plds[ln * 32 + 16 * hi];
            pf.q[0] = *(const u32x4*)pr;
            pf.q[1] = *(const u32x4*)(pr + 8);
        }
        __syncthreads();

        // ---- O += P . V  (V pre-transposed: Vt[d, s] contiguous in s)
#pragma unroll
        for (int dt = 0; dt < 8; ++dt) {
            Frag vf;
            const bf16* vp = vb + (size_t)(dt * 16 + ln) * SEQ + kv0 + 16 * hi;
            vf.q[0] = *(const u32x4*)vp;
            vf.q[1] = *(const u32x4*)(vp + 8);
            oacc[dt] = wmma_bf16(pf, vf, oacc[dt]);
        }
    }

    // ---- normalize and write bf16 output tile [16 x 128]
    float inv[8];
#pragma unroll
    for (int r = 0; r < 8; ++r) inv[r] = 1.0f / lrow[r];
#pragma unroll
    for (int dt = 0; dt < 8; ++dt)
#pragma unroll
        for (int r = 0; r < 8; ++r) {
            int m = qb + r + 8 * hi;
            O[(size_t)(b * SEQ + m) * HID + h * HD + dt * 16 + ln] =
                f2bf(oacc[dt][r] * inv[r]);
        }
}

// ---------------------------------------------------------------------------
// Host launcher
// ---------------------------------------------------------------------------
extern "C" void kernel_launch(void* const* d_in, const int* in_sizes, int n_in,
                              void* d_out, int out_size, void* d_ws, size_t ws_size,
                              hipStream_t stream) {
    const float* x      = (const float*)d_in[0];
    const float* W_qkv  = (const float*)d_in[1];
    const float* b_qkv  = (const float*)d_in[2];
    const float* W_proj = (const float*)d_in[3];
    const float* b_proj = (const float*)d_in[4];
    float* out = (float*)d_out;

    // workspace layout (bytes); xb is reused as the attention-output buffer
    char* ws = (char*)d_ws;
    bf16* xb     = (bf16*)(ws);                        // 4096*2048*2 = 16,777,216
    bf16* wqkvT  = (bf16*)(ws + 16777216);             // 2304*2048*2 =  9,437,184
    bf16* wprojT = (bf16*)(ws + 26214400);             // 2048*2048*2 =  8,388,608
    bf16* qbuf   = (bf16*)(ws + 34603008);             // 4096*2048*2 = 16,777,216
    bf16* kbuf   = (bf16*)(ws + 51380224);             // 4096*128*2  =  1,048,576
    bf16* vtbuf  = (bf16*)(ws + 52428800);             // 2*128*2048*2=  1,048,576
    bf16* attn   = xb;                                 // reuse after QKV GEMM

    // 1) x -> bf16
    cvt_bf16_kernel<<<dim3((MROWS * HID / 4 + 255) / 256), dim3(256), 0, stream>>>(
        x, xb, MROWS * HID / 4);

    // 2) weights -> bf16 transposed
    transpose_cvt_kernel<<<dim3(HID / 32, NQKV / 32), dim3(32, 8), 0, stream>>>(
        W_qkv, wqkvT, HID, NQKV);
    transpose_cvt_kernel<<<dim3(HID / 32, HID / 32), dim3(32, 8), 0, stream>>>(
        W_proj, wprojT, HID, HID);

    // 3) QKV GEMM (M=4096, N=2304, K=2048) with split epilogue
    gemm_wmma_kernel<<<dim3(MROWS / 128, NQKV / 128), dim3(256), 0, stream>>>(
        xb, wqkvT, b_qkv, HID, 0, qbuf, kbuf, vtbuf, nullptr);

    // 4) causal MQA flash attention -> attn (bf16, reuses xb)
    mqa_attn_kernel<<<dim3(SEQ / 16, NHEAD, BATCH), dim3(32), 0, stream>>>(
        qbuf, kbuf, vtbuf, attn);

    // 5) output projection (M=4096, N=2048, K=2048) -> fp32 d_out
    gemm_wmma_kernel<<<dim3(MROWS / 128, HID / 128), dim3(256), 0, stream>>>(
        attn, wprojT, b_proj, HID, 1, nullptr, nullptr, nullptr, out);
}
